// Point_TemporalPriorEncoder_6047313953101
// MI455X (gfx1250) — compile-verified, matrix operand order checked by reading, not run on hardware
//
#include <hip/hip_runtime.h>
#include <hip/hip_bf16.h>
#include <stdint.h>

typedef __attribute__((ext_vector_type(16))) __bf16 v16bf;
typedef __attribute__((ext_vector_type(8)))  __bf16 v8bf;
typedef __attribute__((ext_vector_type(8)))  float  v8f;

#define CDIM 64
#define BLOCK_THREADS 256          // 8 waves
#define ROWS_PER_BLOCK 256         // 8 waves * 2 tiles * 16 rows

// ---------------------------------------------------------------------------
// fp32 -> bf16 convert with a zero pad row appended at row N
// ---------------------------------------------------------------------------
__global__ void cvt_pad_kernel(const float* __restrict__ src, __bf16* __restrict__ dst,
                               long nElems, long totElems) {
    long t = (long)blockIdx.x * blockDim.x + threadIdx.x;
    if (t >= totElems) return;
    dst[t] = (t < nElems) ? (__bf16)src[t] : (__bf16)0.0f;
}

__global__ void zero_row_kernel(__bf16* dst, long rowStartElems) {
    dst[rowStartElems + threadIdx.x] = (__bf16)0.0f;
}

// ---------------------------------------------------------------------------
// Pack W[K][64][64] (fp32, [cin][cout]) into WMMA B-fragment-ready bf16:
// fragment f = (k*2 + cb)*4 + nb  (cb = cin 32-block, nb = cout 16-block)
// element t = f*512 + lane*16 + j ; lane holds col n = nb*16 + (lane&15),
// K elements kb+j (j<8) and kb+16+(j-8) (j>=8), kb = (lane>=16)*8 —
// mirroring the 16-bit A-matrix per-lane K ordering.
// ---------------------------------------------------------------------------
__global__ void pack_w_kernel(const float* __restrict__ W, __bf16* __restrict__ Wp, int total) {
    int t = blockIdx.x * blockDim.x + threadIdx.x;
    if (t >= total) return;
    int j    = t & 15;
    int lane = (t >> 4) & 31;
    int f    = t >> 9;
    int nb   = f & 3;
    int cb   = (f >> 2) & 1;
    int k    = f >> 3;
    int n    = nb * 16 + (lane & 15);
    int kb   = (lane >> 4) * 8;
    int cin  = cb * 32 + kb + ((j < 8) ? j : (j + 8));
    Wp[t] = (__bf16)W[(k * CDIM + cin) * CDIM + n];
}

// ---------------------------------------------------------------------------
// Fused sparse conv: out[m] = sum_k A[in_map[k][m]] @ W[k] + bias
//                    (+ optional gather-residual, optional relu)
// 8 waves/block; each wave computes 2 tiles of 16 rows x 64 cols.
// Weights double-buffered in LDS (8KB/k); A-gather fragments double-buffered
// in two register sets with the k-loop unrolled by 2 (explicit role swap, no
// register copies -> no WMMA->VALU hazard NOPs on the swap). Edge rows clamp
// to the zero pad row (index Nin) so EXEC stays all-ones for WMMA and barrier
// participation is uniform.
// ---------------------------------------------------------------------------
__global__ __launch_bounds__(BLOCK_THREADS)
void sparse_conv_kernel(const __bf16* __restrict__ A,      // (Nin+1) x 64 bf16
                        const __bf16* __restrict__ Wp,     // K * 8 * 512 packed bf16
                        const float*  __restrict__ bias,   // 64
                        const int*    __restrict__ inmap,  // K x M
                        const float*  __restrict__ addsrc, // Nadd x 64 fp32 (or null)
                        const int*    __restrict__ addidx, // M (or null)
                        float*        __restrict__ outf,   // M x 64 fp32 (or null)
                        __bf16*       __restrict__ outb,   // (M+1) x 64 bf16 (or null)
                        int M, int Nin, int Nadd, int K, int relu)
{
    __shared__ __bf16 ldsW[2 * 4096];   // 2 x 8KB double buffer; k -> buffer (k&1)

    const int lane = threadIdx.x & 31;
    const int wave = threadIdx.x >> 5;
    const int r    = lane & 15;
    const int hi   = lane >> 4;
    const int kb   = hi * 8;

    const int m0a = blockIdx.x * ROWS_PER_BLOCK + wave * 32;
    const int m0b = m0a + 16;
    const int mra = m0a + r, mrb = m0b + r;
    const bool oka = (mra < M), okb = (mrb < M);
    const int mca = oka ? mra : (M - 1);
    const int mcb = okb ? mrb : (M - 1);

    v8f accA[4] = {{0.f}, {0.f}, {0.f}, {0.f}};
    v8f accB[4] = {{0.f}, {0.f}, {0.f}, {0.f}};

    // cooperative stage of W[k] into LDS buffer (k&1) (256 threads x 32B)
    auto stage = [&](int k) {
        const __bf16* src = Wp + (long)k * 4096 + threadIdx.x * 16;
        *(v16bf*)&ldsW[(k & 1) * 4096 + threadIdx.x * 16] = *(const v16bf*)src;
    };

    // gather both row-tiles' A fragments for kernel offset k
    auto loadA = [&](int k, v16bf& f0A, v16bf& f1A, v16bf& f0B, v16bf& f1B) {
        int ia = inmap[(long)k * M + mca]; if (!oka) ia = Nin;
        int ib = inmap[(long)k * M + mcb]; if (!okb) ib = Nin;
        const __bf16* ra = A + (long)ia * CDIM;
        const __bf16* rb = A + (long)ib * CDIM;
        v8bf x0 = *(const v8bf*)(ra + kb);
        v8bf x1 = *(const v8bf*)(ra + kb + 16);
        v8bf x2 = *(const v8bf*)(ra + 32 + kb);
        v8bf x3 = *(const v8bf*)(ra + 32 + kb + 16);
        v8bf y0 = *(const v8bf*)(rb + kb);
        v8bf y1 = *(const v8bf*)(rb + kb + 16);
        v8bf y2 = *(const v8bf*)(rb + 32 + kb);
        v8bf y3 = *(const v8bf*)(rb + 32 + kb + 16);
        f0A = __builtin_shufflevector(x0, x1, 0,1,2,3,4,5,6,7,8,9,10,11,12,13,14,15);
        f1A = __builtin_shufflevector(x2, x3, 0,1,2,3,4,5,6,7,8,9,10,11,12,13,14,15);
        f0B = __builtin_shufflevector(y0, y1, 0,1,2,3,4,5,6,7,8,9,10,11,12,13,14,15);
        f1B = __builtin_shufflevector(y2, y3, 0,1,2,3,4,5,6,7,8,9,10,11,12,13,14,15);
    };

    // 16 WMMAs for one k, B fragments pulled in groups of four from LDS
    auto computeK = [&](const v16bf& f0A, const v16bf& f1A,
                        const v16bf& f0B, const v16bf& f1B, int buf) {
        const __bf16* lw = &ldsW[buf * 4096 + lane * 16];
        {
            v16bf w0 = *(const v16bf*)(lw + 0 * 512);
            v16bf w1 = *(const v16bf*)(lw + 1 * 512);
            v16bf w2 = *(const v16bf*)(lw + 2 * 512);
            v16bf w3 = *(const v16bf*)(lw + 3 * 512);
            accA[0] = __builtin_amdgcn_wmma_f32_16x16x32_bf16(false, f0A, false, w0, (short)0, accA[0], false, false);
            accB[0] = __builtin_amdgcn_wmma_f32_16x16x32_bf16(false, f0B, false, w0, (short)0, accB[0], false, false);
            accA[1] = __builtin_amdgcn_wmma_f32_16x16x32_bf16(false, f0A, false, w1, (short)0, accA[1], false, false);
            accB[1] = __builtin_amdgcn_wmma_f32_16x16x32_bf16(false, f0B, false, w1, (short)0, accB[1], false, false);
            accA[2] = __builtin_amdgcn_wmma_f32_16x16x32_bf16(false, f0A, false, w2, (short)0, accA[2], false, false);
            accB[2] = __builtin_amdgcn_wmma_f32_16x16x32_bf16(false, f0B, false, w2, (short)0, accB[2], false, false);
            accA[3] = __builtin_amdgcn_wmma_f32_16x16x32_bf16(false, f0A, false, w3, (short)0, accA[3], false, false);
            accB[3] = __builtin_amdgcn_wmma_f32_16x16x32_bf16(false, f0B, false, w3, (short)0, accB[3], false, false);
        }
        {
            v16bf w0 = *(const v16bf*)(lw + 4 * 512);
            v16bf w1 = *(const v16bf*)(lw + 5 * 512);
            v16bf w2 = *(const v16bf*)(lw + 6 * 512);
            v16bf w3 = *(const v16bf*)(lw + 7 * 512);
            accA[0] = __builtin_amdgcn_wmma_f32_16x16x32_bf16(false, f1A, false, w0, (short)0, accA[0], false, false);
            accB[0] = __builtin_amdgcn_wmma_f32_16x16x32_bf16(false, f1B, false, w0, (short)0, accB[0], false, false);
            accA[1] = __builtin_amdgcn_wmma_f32_16x16x32_bf16(false, f1A, false, w1, (short)0, accA[1], false, false);
            accB[1] = __builtin_amdgcn_wmma_f32_16x16x32_bf16(false, f1B, false, w1, (short)0, accB[1], false, false);
            accA[2] = __builtin_amdgcn_wmma_f32_16x16x32_bf16(false, f1A, false, w2, (short)0, accA[2], false, false);
            accB[2] = __builtin_amdgcn_wmma_f32_16x16x32_bf16(false, f1B, false, w2, (short)0, accB[2], false, false);
            accA[3] = __builtin_amdgcn_wmma_f32_16x16x32_bf16(false, f1A, false, w3, (short)0, accA[3], false, false);
            accB[3] = __builtin_amdgcn_wmma_f32_16x16x32_bf16(false, f1B, false, w3, (short)0, accB[3], false, false);
        }
    };

    // two A-fragment register sets, roles alternate (no copies)
    v16bf p0A = {}, p1A = {}, p0B = {}, p1B = {};   // even k
    v16bf q0A = {}, q1A = {}, q0B = {}, q1B = {};   // odd k

    loadA(0, p0A, p1A, p0B, p1B);
    stage(0);
    __syncthreads();

    int k = 0;
    for (; k + 1 < K; k += 2) {
        // first half: compute k (even, buf0, set P); prefetch k+1 (buf1, set Q)
        stage(k + 1);
        loadA(k + 1, q0A, q1A, q0B, q1B);
        computeK(p0A, p1A, p0B, p1B, 0);
        __syncthreads();

        // second half: compute k+1 (odd, buf1, set Q); prefetch k+2 (buf0, set P)
        if (k + 2 < K) {
            stage(k + 2);
            loadA(k + 2, p0A, p1A, p0B, p1B);
        }
        computeK(q0A, q1A, q0B, q1B, 1);
        __syncthreads();
    }
    if (k < K) {                       // odd K remainder (even k, buf0, set P)
        computeK(p0A, p1A, p0B, p1B, 0);
    }

    // Epilogue. C/D layout: VGPR rr -> row rr + hi*8, col r within 16-tile.
    const float bb0 = bias[r], bb1 = bias[16 + r], bb2 = bias[32 + r], bb3 = bias[48 + r];
    auto epil = [&](int m0, v8f* acc) {
        #pragma unroll
        for (int rr = 0; rr < 8; ++rr) {
            int mm = m0 + rr + hi * 8;
            if (mm >= M) continue;
            float add0 = 0.f, add1 = 0.f, add2 = 0.f, add3 = 0.f;
            if (addidx) {
                int ai = addidx[mm];
                if (ai < Nadd) {
                    const float* srow = addsrc + (long)ai * CDIM + r;
                    add0 = srow[0]; add1 = srow[16]; add2 = srow[32]; add3 = srow[48];
                }
            }
            float v0 = acc[0][rr] + bb0 + add0;
            float v1 = acc[1][rr] + bb1 + add1;
            float v2 = acc[2][rr] + bb2 + add2;
            float v3 = acc[3][rr] + bb3 + add3;
            if (relu) { v0 = fmaxf(v0, 0.f); v1 = fmaxf(v1, 0.f); v2 = fmaxf(v2, 0.f); v3 = fmaxf(v3, 0.f); }
            long base = (long)mm * CDIM + r;
            if (outf) { outf[base] = v0; outf[base + 16] = v1; outf[base + 32] = v2; outf[base + 48] = v3; }
            if (outb) {
                outb[base]      = (__bf16)v0; outb[base + 16] = (__bf16)v1;
                outb[base + 32] = (__bf16)v2; outb[base + 48] = (__bf16)v3;
            }
        }
    };
    epil(m0a, accA);
    epil(m0b, accB);
}

// ---------------------------------------------------------------------------
extern "C" void kernel_launch(void* const* d_in, const int* in_sizes, int n_in,
                              void* d_out, int out_size, void* d_ws, size_t ws_size,
                              hipStream_t stream) {
    const float* feats1 = (const float*)d_in[0];
    const float* feats2 = (const float*)d_in[1];
    const float* feats3 = (const float*)d_in[2];
    const float* W[5]   = {(const float*)d_in[3], (const float*)d_in[5], (const float*)d_in[7],
                           (const float*)d_in[9], (const float*)d_in[11]};
    const float* Bv[5]  = {(const float*)d_in[4], (const float*)d_in[6], (const float*)d_in[8],
                           (const float*)d_in[10], (const float*)d_in[12]};
    const int* in1 = (const int*)d_in[13];
    const int* in2 = (const int*)d_in[15];
    const int* in3 = (const int*)d_in[17];
    const int* in4 = (const int*)d_in[19];
    const int* in5 = (const int*)d_in[21];
    const int* match2 = (const int*)d_in[23];
    const int* match3 = (const int*)d_in[24];

    const int N1  = in_sizes[0] / CDIM;   // 240000 stride-1 coords
    const int M2  = in_sizes[13] / 8;     // 120000
    const int M3  = in_sizes[15] / 8;     // 40000
    const int Mo3 = in_sizes[17] / 8;     // conv3 output rows (== out_size/64)
    const int K45 = in_sizes[19] / Mo3;   // 27
    const int Kc[5] = {8, 8, 8, K45, K45};

    // workspace partition (256B aligned)
    uintptr_t p = (uintptr_t)d_ws;
    auto take = [&](size_t bytes) -> void* {
        uintptr_t q = (p + 255) & ~(uintptr_t)255;
        p = q + bytes;
        return (void*)q;
    };
    __bf16* f1b  = (__bf16*)take((size_t)(N1  + 1) * CDIM * 2);
    __bf16* buf2 = (__bf16*)take((size_t)(M2  + 1) * CDIM * 2);
    __bf16* buf3 = (__bf16*)take((size_t)(M3  + 1) * CDIM * 2);
    __bf16* buf4 = (__bf16*)take((size_t)(Mo3 + 1) * CDIM * 2);
    __bf16* buf5 = (__bf16*)take((size_t)(Mo3 + 1) * CDIM * 2);
    __bf16* Wp[5];
    for (int i = 0; i < 5; ++i) Wp[i] = (__bf16*)take((size_t)Kc[i] * 4096 * 2);

    // stage 0: convert feats1 to bf16 (+pad), zero pad rows, pack weights
    {
        long n = (long)N1 * CDIM, tot = (long)(N1 + 1) * CDIM;
        cvt_pad_kernel<<<(int)((tot + 255) / 256), 256, 0, stream>>>(feats1, f1b, n, tot);
    }
    zero_row_kernel<<<1, 64, 0, stream>>>(buf2, (long)M2  * CDIM);
    zero_row_kernel<<<1, 64, 0, stream>>>(buf3, (long)M3  * CDIM);
    zero_row_kernel<<<1, 64, 0, stream>>>(buf4, (long)Mo3 * CDIM);
    zero_row_kernel<<<1, 64, 0, stream>>>(buf5, (long)Mo3 * CDIM);
    for (int i = 0; i < 5; ++i) {
        int tot = Kc[i] * 4096;
        pack_w_kernel<<<(tot + 255) / 256, 256, 0, stream>>>(W[i], Wp[i], tot);
    }

    auto conv = [&](const __bf16* Ain, int i, const int* map,
                    const float* addsrc, const int* addidx,
                    float* outf, __bf16* outb, int M, int Nin, int Nadd, int relu) {
        int blocks = (M + ROWS_PER_BLOCK - 1) / ROWS_PER_BLOCK;
        sparse_conv_kernel<<<blocks, BLOCK_THREADS, 0, stream>>>(
            Ain, Wp[i], Bv[i], map, addsrc, addidx, outf, outb, M, Nin, Nadd, Kc[i], relu);
    };

    conv(f1b,  0, in1, feats2, match2, nullptr,        buf2,    M2,  N1,  M2, 0); // conv1 + gather(feats2)
    conv(buf2, 1, in2, feats3, match3, nullptr,        buf3,    M3,  M2,  M3, 0); // conv2 + gather(feats3)
    conv(buf3, 2, in3, nullptr, nullptr, nullptr,      buf4,    Mo3, M3,  0,  0); // conv3
    conv(buf4, 3, in4, nullptr, nullptr, nullptr,      buf5,    Mo3, Mo3, 0,  1); // relu(conv4)
    conv(buf5, 4, in5, nullptr, nullptr, (float*)d_out, nullptr, Mo3, Mo3, 0,  0); // conv5 -> d_out fp32
}
